// H2GCN_12008728560067
// MI455X (gfx1250) — matrix operand order repaired and checked by reading, not compiled, not run on hardware
//
#include <hip/hip_runtime.h>
#include <hip/hip_bf16.h>
#include <stdint.h>

#define DD 512
#define NTY 3
#define KK 10
#define NEG_SLOPE 0.2f
#define EPSF 1e-6f

typedef __attribute__((ext_vector_type(16))) __bf16 v16bf;
typedef __attribute__((ext_vector_type(8)))  __bf16 v8bf;
typedef __attribute__((ext_vector_type(8)))  float  v8f;
typedef __attribute__((ext_vector_type(4)))  unsigned v4u;
typedef __attribute__((ext_vector_type(8)))  int      v8i;
typedef __attribute__((ext_vector_type(4)))  int      v4i;

__device__ __forceinline__ unsigned short f2bf(float f) {
    unsigned u = __float_as_uint(f);
    unsigned r = u + 0x7FFFu + ((u >> 16) & 1u);   // round-to-nearest-even
    return (unsigned short)(r >> 16);
}
__device__ __forceinline__ float wred32(float v) {
#pragma unroll
    for (int o = 16; o > 0; o >>= 1) v += __shfl_xor(v, o, 32);
    return v;
}
__device__ __forceinline__ float leaky(float x) { return x > 0.f ? x : NEG_SLOPE * x; }
__device__ __forceinline__ unsigned encf(float f) {
    unsigned u = __float_as_uint(f);
    return (u & 0x80000000u) ? ~u : (u | 0x80000000u);
}
__device__ __forceinline__ float decf(unsigned e) {
    unsigned u = (e & 0x80000000u) ? (e & 0x7FFFFFFFu) : ~e;
    return __uint_as_float(u);
}

// ---- TDM: issue a contiguous 16KB global -> LDS DMA (1-D tile descriptor) --
// D# per CDNA5 ISA: group0 = control/addresses,
// group1 = dims (data_size=3 -> 8B units; tile_dim0 = tensor_dim0 = 2048 = 16KB)
// This toolchain's builtin takes 6 args (g0, g1, g2, g3, extra group, cpol).
__device__ __forceinline__ void tdm_load_16k(const unsigned short* gsrc, unsigned lds_off) {
    unsigned long long ga = (unsigned long long)(uintptr_t)gsrc;
    v4u g0;
    g0[0] = 1u;                                            // count=1, user mode
    g0[1] = lds_off;                                       // lds_addr (bytes)
    g0[2] = (unsigned)ga;                                  // global_addr[31:0]
    g0[3] = (unsigned)((ga >> 32) & 0x01FFFFFFu) | (2u << 30);  // addr[56:32] | type=2
    v8i g1;
    g1[0] = (3 << 16);                 // wg_mask=0, data_size=3 (8B), no flags
    g1[1] = (int)(2048u << 16);        // tensor_dim0[15:0] into bits 63:48
    g1[2] = (int)((2048u >> 16) | (1u << 16)); // tensor_dim0[31:16] | tensor_dim1[15:0]=1
    g1[3] = (int)(2048u << 16);        // tensor_dim1[31:16]=0 | tile_dim0=2048
    g1[4] = 0;                         // tile_dim1=0 (unused), tile_dim2=0
    g1[5] = 2048;                      // tensor_dim0_stride[31:0]
    g1[6] = 0;                         // stride hi / dim1 stride lo
    g1[7] = 0;
    v4i z4; z4[0] = 0; z4[1] = 0; z4[2] = 0; z4[3] = 0;    // <=2-D: groups 2/3 unused
    v8i z8;
#pragma unroll
    for (int i = 0; i < 8; ++i) z8[i] = 0;
    __builtin_amdgcn_tensor_load_to_lds(g0, g1, z4, z4, z8, 0);
}

// ---- pack W[512][512] -> bf16 B fragments, layout [kt][nt][lane][16] -------
// B 32x16 bf16 fragment: lane l -> N = nt*16 + (l&15);
// element i (0..15) -> K = kb + 8*(l>>4) + (i&7) + 16*(i>>3)
__global__ void k_pack_w(const float* __restrict__ W, unsigned short* __restrict__ Wp) {
    int tid  = blockIdx.x * blockDim.x + threadIdx.x;   // 0 .. 512*512-1
    int i    = tid & 15;
    int lane = (tid >> 4) & 31;
    int nt   = (tid >> 9) & 31;
    int kt   = tid >> 14;
    int n = nt * 16 + (lane & 15);
    int k = kt * 32 + (lane >> 4) * 8 + (i & 7) + (i >> 3) * 16;
    Wp[tid] = f2bf(W[k * DD + n]);
}

// ---- p / (||p|| + eps) -----------------------------------------------------
__global__ void k_pnorm(const float* __restrict__ p, float* __restrict__ pn) {
    __shared__ float sh[DD];
    int t = threadIdx.x;
    float v = p[t];
    sh[t] = v * v;
    __syncthreads();
    for (int o = 256; o > 0; o >>= 1) { if (t < o) sh[t] += sh[t + o]; __syncthreads(); }
    float nrm = sqrtf(sh[0]) + EPSF;
    pn[t] = v / nrm;
}

// ---- layernorm(x) -> bf16, one wave per node -------------------------------
__global__ void k_ln_bf16(const float* __restrict__ x, const float* __restrict__ g,
                          const float* __restrict__ bta, unsigned short* __restrict__ out,
                          int Nn) {
    int node = blockIdx.x * 8 + (threadIdx.x >> 5);
    if (node >= Nn) return;
    int lane = threadIdx.x & 31;
    const float* row = x + (size_t)node * DD;
    float vals[16]; float s = 0.f, ss = 0.f;
#pragma unroll
    for (int i = 0; i < 16; ++i) { float v = row[lane + i * 32]; vals[i] = v; s += v; ss += v * v; }
    s = wred32(s); ss = wred32(ss);
    float mean = s * (1.f / DD);
    float var  = ss * (1.f / DD) - mean * mean;
    float r = rsqrtf(var + 1e-5f);
    unsigned short* orow = out + (size_t)node * DD;
#pragma unroll
    for (int i = 0; i < 16; ++i) {
        int c = lane + i * 32;
        orow[c] = f2bf((vals[i] - mean) * r * g[c] + bta[c]);
    }
}

// ---- WMMA GEMM with TDM-staged, double-buffered B in LDS -------------------
// block = 256 thr = 8 waves (2 M-tiles x 4 N-waves); wave tile = 16M x 64N.
// Per kt (K-step of 32): B slab for this block's 256 columns = 16KB contiguous
// in Wp; wave 0 DMAs it into LDS via tensor_load_to_lds while other buffer is
// consumed; A fragments stream from global.
__global__ void __launch_bounds__(256) k_gemm_wmma(
    const unsigned short* __restrict__ A, const unsigned short* __restrict__ Bp,
    const float* __restrict__ bias, float* __restrict__ C, int M) {
    __shared__ unsigned short Bs[2][8192];     // 2 x 16KB
    int lane = threadIdx.x & 31;
    int w    = threadIdx.x >> 5;
    int mbase = blockIdx.y * 32 + (w >> 2) * 16;
    int nh    = blockIdx.x;                    // 0..1 -> which 256-col half
    int nbl   = (w & 3) * 4;                   // wave's local n-tile base (of 16)
    bool active = (mbase < M);                 // M multiple of 16
    int mrow = active ? (mbase + (lane & 15)) : (lane & 15);
    int hh = lane >> 4;
    const __bf16* arow = (const __bf16*)(A + (size_t)mrow * DD);

    if (threadIdx.x < 32)                       // wave 0: prefetch kt=0 slab
        tdm_load_16k(Bp + (size_t)nh * 8192, 0u);

    v8f acc0 = {}, acc1 = {}, acc2 = {}, acc3 = {};
    for (int kt = 0; kt < 16; ++kt) {
        __syncthreads();                        // readers of buffer being overwritten done
        if (threadIdx.x < 32) {
            if (kt < 15) {
                tdm_load_16k(Bp + (size_t)(kt + 1) * 16384 + (size_t)nh * 8192,
                             ((kt + 1) & 1) ? 16384u : 0u);
                __builtin_amdgcn_s_wait_tensorcnt((short)1);   // kt slab landed
            } else {
                __builtin_amdgcn_s_wait_tensorcnt((short)0);
            }
        }
        __syncthreads();                        // all waves: buf[kt&1] ready
        int kb = kt * 32;
        // A 16-bit 16x32 layout: lanes 0-15 K[0..7]+[16..23], lanes 16-31 K[8..15]+[24..31]
        v8bf lo = *(const v8bf*)(arow + kb + hh * 8);
        v8bf hi = *(const v8bf*)(arow + kb + 16 + hh * 8);
        v16bf a = __builtin_shufflevector(lo, hi, 0,1,2,3,4,5,6,7,8,9,10,11,12,13,14,15);
        const unsigned short* bb = &Bs[kt & 1][0];
        v16bf b0 = *(const v16bf*)(bb + (nbl + 0) * 512 + lane * 16);
        v16bf b1 = *(const v16bf*)(bb + (nbl + 1) * 512 + lane * 16);
        v16bf b2 = *(const v16bf*)(bb + (nbl + 2) * 512 + lane * 16);
        v16bf b3 = *(const v16bf*)(bb + (nbl + 3) * 512 + lane * 16);
        acc0 = __builtin_amdgcn_wmma_f32_16x16x32_bf16(false, a, false, b0, (short)0, acc0, false, false);
        acc1 = __builtin_amdgcn_wmma_f32_16x16x32_bf16(false, a, false, b1, (short)0, acc1, false, false);
        acc2 = __builtin_amdgcn_wmma_f32_16x16x32_bf16(false, a, false, b2, (short)0, acc2, false, false);
        acc3 = __builtin_amdgcn_wmma_f32_16x16x32_bf16(false, a, false, b3, (short)0, acc3, false, false);
    }
    if (!active) return;
    // C/D layout: VGPR r, lane l -> M = r + 8*(l>=16), N = l&15
    int nbase = nh * 256 + (w & 3) * 64;
    int nofs = lane & 15;
    float b0v = bias[nbase +  0 + nofs];
    float b1v = bias[nbase + 16 + nofs];
    float b2v = bias[nbase + 32 + nofs];
    float b3v = bias[nbase + 48 + nofs];
#pragma unroll
    for (int r = 0; r < 8; ++r) {
        size_t mm = (size_t)(mbase + hh * 8 + r) * DD;
        C[mm + nbase +  0 + nofs] = acc0[r] + b0v;
        C[mm + nbase + 16 + nofs] = acc1[r] + b1v;
        C[mm + nbase + 32 + nofs] = acc2[r] + b2v;
        C[mm + nbase + 48 + nofs] = acc3[r] + b3v;
    }
}

// ---- per-node dot products h.a_src, h.a_dst, h.t_dst -----------------------
__global__ void k_node_dots(const float* __restrict__ h, const float* __restrict__ asrc,
                            const float* __restrict__ adst, const float* __restrict__ tdst,
                            float* __restrict__ dA, float* __restrict__ dB,
                            float* __restrict__ dT, int Nn) {
    int node = blockIdx.x * 8 + (threadIdx.x >> 5);
    if (node >= Nn) return;
    int lane = threadIdx.x & 31;
    const float* row = h + (size_t)node * DD;
    float s0 = 0.f, s1 = 0.f, s2 = 0.f;
#pragma unroll
    for (int i = 0; i < 16; ++i) {
        int c = lane + i * 32; float v = row[c];
        s0 += v * asrc[c]; s1 += v * adst[c]; s2 += v * tdst[c];
    }
    s0 = wred32(s0); s1 = wred32(s1); s2 = wred32(s2);
    if (lane == 0) { dA[node] = s0; dB[node] = s1; dT[node] = s2; }
}

// ---- edge logits + segment max ---------------------------------------------
__global__ void k_edge_logit(const int* __restrict__ src, const int* __restrict__ dst,
                             const int* __restrict__ ntype, const float* __restrict__ dA,
                             const float* __restrict__ dB, float* __restrict__ elog,
                             unsigned* __restrict__ segmax, int E) {
    int e = blockIdx.x * blockDim.x + threadIdx.x;
    if (e >= E) return;
    int s = src[e], d = dst[e];
    float l = leaky(dB[d] + dA[s]);
    elog[e] = l;
    atomicMax(&segmax[d * NTY + ntype[s]], encf(l));
}

// ---- edge exp + segment sum + count ----------------------------------------
__global__ void k_edge_exp(const int* __restrict__ src, const int* __restrict__ dst,
                           const int* __restrict__ ntype, float* __restrict__ elog,
                           const unsigned* __restrict__ segmax, float* __restrict__ segsum,
                           float* __restrict__ cnt, int E) {
    int e = blockIdx.x * blockDim.x + threadIdx.x;
    if (e >= E) return;
    int s = src[e], d = dst[e];
    int seg = d * NTY + ntype[s];
    float ev = __expf(elog[e] - decf(segmax[seg]));
    elog[e] = ev;                          // reuse buffer for exp values
    atomicAdd(&segsum[seg], ev);
    atomicAdd(&cnt[seg], 1.0f);
}

// ---- weighted message scatter: m[seg] += attn * h[src], wave per edge ------
__global__ void k_edge_agg(const int* __restrict__ src, const int* __restrict__ dst,
                           const int* __restrict__ ntype, const float* __restrict__ ev,
                           const float* __restrict__ segsum, const float* __restrict__ h,
                           float* __restrict__ m, int E) {
    int e = blockIdx.x * 8 + (threadIdx.x >> 5);
    if (e >= E) return;
    int lane = threadIdx.x & 31;
    int s = src[e], d = dst[e];
    int seg = d * NTY + ntype[s];
    float attn = ev[e] / (segsum[seg] + 1e-16f);
    const float* hr = h + (size_t)s * DD;
    float* mr = m + (size_t)seg * DD;
#pragma unroll
    for (int i = 0; i < 16; ++i) { int c = lane + i * 32; atomicAdd(&mr[c], attn * hr[c]); }
}

// ---- per-node: resolution softmax, combine, layernorm, score ---------------
__global__ void k_combine(const float* __restrict__ m, const float* __restrict__ dT,
                          const float* __restrict__ cnt, const float* __restrict__ tsrc,
                          const float* __restrict__ g, const float* __restrict__ bta,
                          const float* __restrict__ pn, float* __restrict__ hfin,
                          float* __restrict__ score, float* __restrict__ swork, int Nn) {
    int node = blockIdx.x * 8 + (threadIdx.x >> 5);
    if (node >= Nn) return;
    int lane = threadIdx.x & 31;
    const float* mb = m + (size_t)node * NTY * DD;
    float mv[NTY][16];
    float dot[NTY];
#pragma unroll
    for (int t = 0; t < NTY; ++t) {
        float s = 0.f;
#pragma unroll
        for (int i = 0; i < 16; ++i) {
            int c = lane + i * 32; float v = mb[t * DD + c];
            mv[t][i] = v; s += v * tsrc[c];
        }
        dot[t] = wred32(s);
    }
    float td = dT[node];
    float bl[NTY];
#pragma unroll
    for (int t = 0; t < NTY; ++t) {
        float l = leaky(td + dot[t]);
        bl[t] = (cnt[node * NTY + t] > 0.f) ? l : -1e9f;
    }
    float mx = fmaxf(bl[0], fmaxf(bl[1], bl[2]));
    float e0 = __expf(bl[0] - mx), e1 = __expf(bl[1] - mx), e2 = __expf(bl[2] - mx);
    float inv = 1.f / (e0 + e1 + e2);
    float w0 = e0 * inv, w1 = e1 * inv, w2 = e2 * inv;
    float out[16]; float s = 0.f, ss = 0.f;
#pragma unroll
    for (int i = 0; i < 16; ++i) {
        float v = w0 * mv[0][i] + w1 * mv[1][i] + w2 * mv[2][i];
        out[i] = v; s += v; ss += v * v;
    }
    s = wred32(s); ss = wred32(ss);
    float mean = s * (1.f / DD);
    float var  = ss * (1.f / DD) - mean * mean;
    float r = rsqrtf(var + 1e-5f);
    float* orow = hfin + (size_t)node * DD;
    float sd = 0.f;
#pragma unroll
    for (int i = 0; i < 16; ++i) {
        int c = lane + i * 32;
        float v = (out[i] - mean) * r * g[c] + bta[c];
        orow[c] = v;
        sd += v * pn[c];
    }
    sd = wred32(sd);
    if (lane == 0) {
        float sc = 1.f / (1.f + __expf(-sd));
        score[node] = sc; swork[node] = sc;
    }
}

// ---- iterative top-10 selection (single block) -----------------------------
__global__ void k_topk(float* __restrict__ sw, const float* __restrict__ xy,
                       int* __restrict__ centers, float* __restrict__ cpos, int Nn) {
    __shared__ float sv[256];
    __shared__ int   si[256];
    int t = threadIdx.x;
    for (int k = 0; k < KK; ++k) {
        float best = -3.4e38f; int bi = 0x7fffffff;
        for (int i = t; i < Nn; i += 256) {
            float v = sw[i];
            if (v > best || (v == best && i < bi)) { best = v; bi = i; }
        }
        sv[t] = best; si[t] = bi;
        __syncthreads();
        for (int o = 128; o > 0; o >>= 1) {
            if (t < o) {
                if (sv[t + o] > sv[t] || (sv[t + o] == sv[t] && si[t + o] < si[t])) {
                    sv[t] = sv[t + o]; si[t] = si[t + o];
                }
            }
            __syncthreads();
        }
        if (t == 0) {
            int c = si[0];
            centers[k]     = c;
            cpos[2 * k]     = xy[2 * c] * 2.f - 1.f;
            cpos[2 * k + 1] = xy[2 * c + 1] * 2.f - 1.f;
            sw[c] = -3.4e38f;
        }
        __syncthreads();
    }
}

// ---- nearest-center assignment + score-gated pooling, wave per node --------
__global__ void k_pool(const float* __restrict__ xy, const float* __restrict__ cpos,
                       const float* __restrict__ score, const float* __restrict__ hfin,
                       float* __restrict__ num, float* __restrict__ den, int Nn) {
    int node = blockIdx.x * 8 + (threadIdx.x >> 5);
    if (node >= Nn) return;
    int lane = threadIdx.x & 31;
    float px = xy[node * 2] * 2.f - 1.f;
    float py = xy[node * 2 + 1] * 2.f - 1.f;
    int cl = 0; float bd = 3.4e38f;
#pragma unroll
    for (int k = 0; k < KK; ++k) {
        float dx = px - cpos[2 * k], dy = py - cpos[2 * k + 1];
        float d2 = dx * dx + dy * dy;
        if (d2 < bd) { bd = d2; cl = k; }
    }
    float sc = score[node];
    const float* hr = hfin + (size_t)node * DD;
    float* nr = num + (size_t)cl * DD;
#pragma unroll
    for (int i = 0; i < 16; ++i) { int c = lane + i * 32; atomicAdd(&nr[c], sc * hr[c]); }
    if (lane == 0) atomicAdd(&den[cl], sc);
}

// ---- finalize: out = num / (den + eps) -------------------------------------
__global__ void k_final(const float* __restrict__ num, const float* __restrict__ den,
                        float* __restrict__ out) {
    int idx = blockIdx.x * blockDim.x + threadIdx.x;
    if (idx >= KK * DD) return;
    out[idx] = num[idx] / (den[idx >> 9] + EPSF);
}

extern "C" void kernel_launch(void* const* d_in, const int* in_sizes, int n_in,
                              void* d_out, int out_size, void* d_ws, size_t ws_size,
                              hipStream_t stream) {
    const float* x     = (const float*)d_in[0];
    const int*   ei    = (const int*)d_in[1];
    const int*   ntype = (const int*)d_in[2];
    /* d_in[3] = tree (unused for new_x output) */
    const float* xy    = (const float*)d_in[4];
    const float* W     = (const float*)d_in[5];
    const float* bias  = (const float*)d_in[6];
    const float* asrc  = (const float*)d_in[7];
    const float* adst  = (const float*)d_in[8];
    const float* tsrc  = (const float*)d_in[9];
    const float* tdst  = (const float*)d_in[10];
    const float* p     = (const float*)d_in[11];
    const float* g     = (const float*)d_in[12];
    const float* bta   = (const float*)d_in[13];

    const int Nn = in_sizes[0] / DD;     // 30000
    const int E  = in_sizes[1] / 2;      // 480000
    const int* src = ei;
    const int* dst = ei + E;

    char* ws = (char*)d_ws;
    size_t off = 0;
    auto carve = [&](size_t bytes) -> char* {
        char* ptr = ws + off;
        off += (bytes + 255) & ~(size_t)255;
        return ptr;
    };
    unsigned short* h0b  = (unsigned short*)carve((size_t)Nn * DD * 2);
    unsigned short* Wp   = (unsigned short*)carve((size_t)DD * DD * 2);
    float*    h1     = (float*)carve((size_t)Nn * DD * 4);   // GEMM out; reused as hfin
    float*    m      = (float*)carve((size_t)Nn * NTY * DD * 4);
    float*    dA     = (float*)carve((size_t)Nn * 4);
    float*    dB     = (float*)carve((size_t)Nn * 4);
    float*    dT     = (float*)carve((size_t)Nn * 4);
    unsigned* segmax = (unsigned*)carve((size_t)Nn * NTY * 4);
    float*    segsum = (float*)carve((size_t)Nn * NTY * 4);
    float*    cnt    = (float*)carve((size_t)Nn * NTY * 4);
    float*    elog   = (float*)carve((size_t)E * 4);
    float*    score  = (float*)carve((size_t)Nn * 4);
    float*    swork  = (float*)carve((size_t)Nn * 4);
    float*    pn     = (float*)carve((size_t)DD * 4);
    int*      centers= (int*)carve(KK * 4);
    float*    cpos   = (float*)carve(KK * 2 * 4);
    float*    num    = (float*)carve((size_t)KK * DD * 4);
    float*    den    = (float*)carve((size_t)KK * 4);
    float*    hfin   = h1;   // safe alias: k_combine reads only m/dT/cnt, writes hfin

    (void)hipMemsetAsync(segmax, 0, (size_t)Nn * NTY * 4, stream);   // 0u == encoded -inf
    (void)hipMemsetAsync(segsum, 0, (size_t)Nn * NTY * 4, stream);
    (void)hipMemsetAsync(cnt,    0, (size_t)Nn * NTY * 4, stream);
    (void)hipMemsetAsync(m,      0, (size_t)Nn * NTY * DD * 4, stream);
    (void)hipMemsetAsync(num,    0, (size_t)KK * DD * 4, stream);
    (void)hipMemsetAsync(den,    0, (size_t)KK * 4, stream);

    k_pack_w <<<(DD * DD) / 256, 256, 0, stream>>>(W, Wp);
    k_pnorm  <<<1, DD, 0, stream>>>(p, pn);
    k_ln_bf16<<<(Nn + 7) / 8, 256, 0, stream>>>(x, g, bta, h0b, Nn);

    dim3 gg(DD / 256, (Nn + 31) / 32);
    k_gemm_wmma<<<gg, 256, 0, stream>>>(h0b, Wp, bias, h1, Nn);

    k_node_dots <<<(Nn + 7) / 8, 256, 0, stream>>>(h1, asrc, adst, tdst, dA, dB, dT, Nn);
    k_edge_logit<<<(E + 255) / 256, 256, 0, stream>>>(src, dst, ntype, dA, dB, elog, segmax, E);
    k_edge_exp  <<<(E + 255) / 256, 256, 0, stream>>>(src, dst, ntype, elog, segmax, segsum, cnt, E);
    k_edge_agg  <<<(E + 7) / 8, 256, 0, stream>>>(src, dst, ntype, elog, segsum, h1, m, E);
    k_combine   <<<(Nn + 7) / 8, 256, 0, stream>>>(m, dT, cnt, tsrc, g, bta, pn, hfin, score, swork, Nn);
    k_topk      <<<1, 256, 0, stream>>>(swork, xy, centers, cpos, Nn);
    k_pool      <<<(Nn + 7) / 8, 256, 0, stream>>>(xy, cpos, score, hfin, num, den, Nn);
    k_final     <<<(KK * DD + 255) / 256, 256, 0, stream>>>(num, den, (float*)d_out);
}
